// EdgePred_32564442038490
// MI455X (gfx1250) — matrix operand order repaired and checked by reading, not compiled, not run on hardware
//
#include <hip/hip_runtime.h>
#include <hip/hip_bf16.h>

// Inputs (setup_inputs order):
// 0: h    [N_NODES*64] f32
// 1: u    [E] i32
// 2: v    [E] i32
// 3: W1_w [64*64] f32   4: W1_b [64] f32
// 5: W2_w [64*64] f32   6: W2_b [64] f32
// 7: t    [1] f32
// out: [E] f32

typedef __attribute__((ext_vector_type(16))) __bf16 v16bf;
typedef __attribute__((ext_vector_type(8)))  float  v8f;
typedef __attribute__((ext_vector_type(4)))  float  v4f;
typedef __attribute__((ext_vector_type(4)))  int    v4i;

struct FragBits { v4i lo, hi; };   // 32 bytes == one v16bf fragment

union FragWords { v16bf f; int w[8]; };

// --- A-tile fragment: 16x32 bf16 with ReLU.
// ISA 16-bit A layout: lane holds row M=lane%16, K = kb+0..7 (elems 0..7) and
// K = kb+16..23 (elems 8..15), kb = 8*(lane>>4). p pre-offset to row*64+chunk*32+kb.
// Convert f32->bf16 (v_cvt_pk_bf16_f32), then clamp with packed
// v_pk_max_num_bf16 (one op per 2 elems; ReLU commutes with monotone rounding).
__device__ __forceinline__ v16bf load_relu_A(const float* __restrict__ p) {
  v4f x0 = *(const v4f*)(p + 0);
  v4f x1 = *(const v4f*)(p + 4);
  v4f x2 = *(const v4f*)(p + 16);
  v4f x3 = *(const v4f*)(p + 20);
  FragWords u;
#pragma unroll
  for (int i = 0; i < 4; ++i) {
    u.f[i +  0] = (__bf16)x0[i];
    u.f[i +  4] = (__bf16)x1[i];
    u.f[i +  8] = (__bf16)x2[i];
    u.f[i + 12] = (__bf16)x3[i];
  }
#pragma unroll
  for (int i = 0; i < 8; ++i)
    asm("v_pk_max_num_bf16 %0, %1, 0" : "=v"(u.w[i]) : "v"(u.w[i]));
  return u.f;
}

// --- B-tile fragment from global f32 weights: 32x16 bf16. ISA B layout:
// lane holds col N=lane%16, K = kb+0..7 (elems 0..7) and kb+16..23 (elems 8..15), kb = 8*(lane>>4).
// p = W + N*64 + chunk*32 + 8*(lane>>4).
__device__ __forceinline__ v16bf make_B(const float* __restrict__ p) {
  v16bf b;
#pragma unroll
  for (int q = 0; q < 4; ++q) {
    v4f x = *(const v4f*)(p + 4 * q + (q >> 1) * 8);
#pragma unroll
    for (int i = 0; i < 4; ++i) b[4 * q + i] = (__bf16)x[i];
  }
  return b;
}

// --- Load two B fragments (kc=0,1 of one N-tile) from the LDS weight cache.
// asm volatile + "memory" keeps these inside the loop (plain loads would be
// LICM-hoisted back into 128 VGPRs, recreating the register-pressure problem).
__device__ __forceinline__ void lds_load_2frags(unsigned off0, unsigned off1,
                                                v16bf& f0, v16bf& f1) {
  v4i lo0, hi0, lo1, hi1;
  asm volatile("ds_load_b128 %0, %4\n\t"
               "ds_load_b128 %1, %4 offset:16\n\t"
               "ds_load_b128 %2, %5\n\t"
               "ds_load_b128 %3, %5 offset:16\n\t"
               "s_wait_dscnt 0x0"
               : "=v"(lo0), "=v"(hi0), "=v"(lo1), "=v"(hi1)
               : "v"(off0), "v"(off1)
               : "memory");
  f0 = __builtin_bit_cast(v16bf, FragBits{lo0, hi0});
  f1 = __builtin_bit_cast(v16bf, FragBits{lo1, hi1});
}

__global__ __launch_bounds__(256)
void edgepred_wmma_bf16(const float* __restrict__ h,
                        const int*   __restrict__ eu,
                        const int*   __restrict__ ev,
                        const float* __restrict__ W1w,
                        const float* __restrict__ W1b,
                        const float* __restrict__ W2w,
                        const float* __restrict__ W2b,
                        const float* __restrict__ tptr,
                        float*       __restrict__ out,
                        int nEdges) {
  const int lane = threadIdx.x & 31;
  const int half = lane >> 4;   // 0 or 1
  const int lid  = lane & 15;   // 0..15
  const int kbA  = half * 8;    // A-fragment K base within a 32-chunk
  const int kbB  = half * 8;    // B-fragment K base within a 32-chunk

  const int waveInBlk = threadIdx.x >> 5;
  const int waveId    = blockIdx.x * (blockDim.x >> 5) + waveInBlk;
  const int nWaves    = gridDim.x * (blockDim.x >> 5);

  // ---- LDS weight cache: 16 tiles (layer*8 + n*2 + kc), each 32 lanes x 32B
  // bf16 fragment, pre-swizzled into the WMMA B lane layout. 16 KB total.
  __shared__ __attribute__((aligned(32))) v16bf Bfrag[16][32];

  for (int tIdx = waveInBlk; tIdx < 16; tIdx += 8) {   // 8 waves, 2 tiles each
    const int layer = tIdx >> 3;
    const int n     = (tIdx >> 1) & 3;
    const int kc    = tIdx & 1;
    const float* W  = layer ? W2w : W1w;
    const int    N  = n * 16 + lid;
    Bfrag[tIdx][lane] = make_B(W + (size_t)N * 64 + kc * 32 + kbB);
  }
  __syncthreads();

  const unsigned ldsBase = (unsigned)(size_t)&Bfrag[0][lane];

  // Per-lane biases: depend only on the output column N = n*16+lid.
  float b1[4], b2[4];
#pragma unroll
  for (int n = 0; n < 4; ++n) {
    b1[n] = W1b[n * 16 + lid];
    b2[n] = W2b[n * 16 + lid];
  }
  const float t   = tptr[0];
  const float eps = 1e-12f;

  // After the fold-and-route reduction, lane lid holds edge-row
  // j = b3 | (b2<<1) | (b1<<2) (replicated across bit0).
  const int jmap = ((lid >> 3) & 1) | ((lid & 4) >> 1) | ((lid & 2) << 1);

  const int nGroups = (nEdges + 15) >> 4;
  for (int g = waveId; g < nGroups; g += nWaves) {
    const int ebase = g * 16;
    int myEdge = ebase + lid;                  // A-row this lane serves
    if (myEdge >= nEdges) myEdge = nEdges - 1; // clamp tail: EXEC stays uniform
    const float* pu = h + (size_t)eu[myEdge] * 64;
    const float* pv = h + (size_t)ev[myEdge] * 64;

    // ---- Layer 1 on gathered u-rows: cu[n] = relu(h[u]) @ W1^T + b1
    v16bf Au0 = load_relu_A(pu + kbA);
    v16bf Au1 = load_relu_A(pu + 32 + kbA);
    v8f cu[4];
#pragma unroll
    for (int n = 0; n < 4; ++n) {
      v16bf Bk0, Bk1;
      const unsigned off = ldsBase + (unsigned)((0 * 8 + n * 2) * 32 * 32);
      lds_load_2frags(off, off + 32 * 32, Bk0, Bk1);
      v8f c;
#pragma unroll
      for (int j = 0; j < 8; ++j) c[j] = b1[n];
      c = __builtin_amdgcn_wmma_f32_16x16x32_bf16(false, Au0, false, Bk0,
                                                  (short)0, c, false, false);
      c = __builtin_amdgcn_wmma_f32_16x16x32_bf16(false, Au1, false, Bk1,
                                                  (short)0, c, false, false);
      cu[n] = c;
    }

    // ---- Layer 2 on gathered v-rows: cv[n] = relu(h[v]) @ W2^T + b2
    v16bf Av0 = load_relu_A(pv + kbA);
    v16bf Av1 = load_relu_A(pv + 32 + kbA);
    v8f cv[4];
#pragma unroll
    for (int n = 0; n < 4; ++n) {
      v16bf Bk0, Bk1;
      const unsigned off = ldsBase + (unsigned)((1 * 8 + n * 2) * 32 * 32);
      lds_load_2frags(off, off + 32 * 32, Bk0, Bk1);
      v8f c;
#pragma unroll
      for (int j = 0; j < 8; ++j) c[j] = b2[n];
      c = __builtin_amdgcn_wmma_f32_16x16x32_bf16(false, Av0, false, Bk0,
                                                  (short)0, c, false, false);
      c = __builtin_amdgcn_wmma_f32_16x16x32_bf16(false, Av1, false, Bk1,
                                                  (short)0, c, false, false);
      cv[n] = c;
    }

    // ---- Per-lane partials over this lane's 4 columns {lid, 16+lid, 32+lid, 48+lid}.
    float su[8], sv[8], uv[8];
#pragma unroll
    for (int j = 0; j < 8; ++j) {
      float a = 0.f, b = 0.f, c = 0.f;
#pragma unroll
      for (int n = 0; n < 4; ++n) {
        const float x = cu[n][j], y = cv[n][j];
        a = fmaf(x, x, a);
        b = fmaf(y, y, b);
        c = fmaf(x, y, c);
      }
      su[j] = a; sv[j] = b; uv[j] = c;
    }

    // ---- Fold-and-route reduction over the 16 lanes of each half:
    // xor-fold, then pack pairs of edge-rows into lane subgroups so the value
    // count halves with the width. 15 shfl + 15 add + 7 sel per quantity.
    const bool s3 = (lid & 8) != 0;
    const bool s2 = (lid & 4) != 0;
    const bool s1 = (lid & 2) != 0;
#pragma unroll
    for (int j = 0; j < 8; ++j) {
      su[j] += __shfl_xor(su[j], 8, 32);
      sv[j] += __shfl_xor(sv[j], 8, 32);
      uv[j] += __shfl_xor(uv[j], 8, 32);
    }
#pragma unroll
    for (int j = 0; j < 4; ++j) {
      su[j] = s3 ? su[2 * j + 1] : su[2 * j];
      sv[j] = s3 ? sv[2 * j + 1] : sv[2 * j];
      uv[j] = s3 ? uv[2 * j + 1] : uv[2 * j];
    }
#pragma unroll
    for (int j = 0; j < 4; ++j) {
      su[j] += __shfl_xor(su[j], 4, 32);
      sv[j] += __shfl_xor(sv[j], 4, 32);
      uv[j] += __shfl_xor(uv[j], 4, 32);
    }
#pragma unroll
    for (int j = 0; j < 2; ++j) {
      su[j] = s2 ? su[2 * j + 1] : su[2 * j];
      sv[j] = s2 ? sv[2 * j + 1] : sv[2 * j];
      uv[j] = s2 ? uv[2 * j + 1] : uv[2 * j];
    }
#pragma unroll
    for (int j = 0; j < 2; ++j) {
      su[j] += __shfl_xor(su[j], 2, 32);
      sv[j] += __shfl_xor(sv[j], 2, 32);
      uv[j] += __shfl_xor(uv[j], 2, 32);
    }
    float fsu = s1 ? su[1] : su[0];
    float fsv = s1 ? sv[1] : sv[0];
    float fuv = s1 ? uv[1] : uv[0];
    fsu += __shfl_xor(fsu, 1, 32);
    fsv += __shfl_xor(fsv, 1, 32);
    fuv += __shfl_xor(fuv, 1, 32);

    // ---- Final math once per lane (this lane owns edge jmap + 8*half).
    const float du = fmaxf(sqrtf(fsu), eps);
    const float dv = fmaxf(sqrtf(fsv), eps);
    const float score = fuv / (du * dv);
    const float res = 1.0f / (1.0f + __expf(-t * score));

    const int oe = ebase + half * 8 + jmap;
    if (((lid & 1) == 0) && oe < nEdges) out[oe] = res;
  }
}

extern "C" void kernel_launch(void* const* d_in, const int* in_sizes, int n_in,
                              void* d_out, int out_size, void* d_ws, size_t ws_size,
                              hipStream_t stream) {
  const float* h   = (const float*)d_in[0];
  const int*   eu  = (const int*)  d_in[1];
  const int*   ev  = (const int*)  d_in[2];
  const float* W1w = (const float*)d_in[3];
  const float* W1b = (const float*)d_in[4];
  const float* W2w = (const float*)d_in[5];
  const float* W2b = (const float*)d_in[6];
  const float* t   = (const float*)d_in[7];
  float* out = (float*)d_out;

  const int nEdges = in_sizes[1];

  const int threads = 256;   // 8 waves / block (wave32)
  const int blocks  = 640;   // grid-stride over 62500 edge groups
  edgepred_wmma_bf16<<<blocks, threads, 0, stream>>>(
      h, eu, ev, W1w, W1b, W2w, W2b, t, out, nEdges);
}
